// SemanticExtractor_19361712571266
// MI455X (gfx1250) — matrix operand order repaired and checked by reading, not compiled, not run on hardware
//
#include <hip/hip_runtime.h>
#include <cstdint>
#include <cstddef>

// ---------------------------------------------------------------------------
// Types / helpers for CDNA5 (gfx1250) wave32 WMMA
// ---------------------------------------------------------------------------
typedef __bf16 bf16;
typedef __bf16 v16bf __attribute__((ext_vector_type(16)));
typedef __bf16 v8bf  __attribute__((ext_vector_type(8)));
typedef float  v8f   __attribute__((ext_vector_type(8)));

#define DEV static __device__ __forceinline__

DEV bf16 f2bf(float f) {
  unsigned u = __builtin_bit_cast(unsigned, f);
  unsigned r = (u + 0x7FFFu + ((u >> 16) & 1u)) >> 16;
  return __builtin_bit_cast(bf16, (unsigned short)r);
}
DEV float bf2f(bf16 h) {
  unsigned u = (unsigned)__builtin_bit_cast(unsigned short, h) << 16;
  return __builtin_bit_cast(float, u);
}
DEV bf16 bfzero() { return __builtin_bit_cast(bf16, (unsigned short)0); }

DEV v8f vzero8() {
  v8f a;
#pragma unroll
  for (int j = 0; j < 8; ++j) a[j] = 0.0f;
  return a;
}

DEV v16bf comb(v8bf a, v8bf b) {
  return __builtin_shufflevector(a, b, 0, 1, 2, 3, 4, 5, 6, 7,
                                 8, 9, 10, 11, 12, 13, 14, 15);
}

// A-matrix fragment (16x32 bf16, MxK): lane holds row m = mbase + (lane&15);
// element i maps to k = (i>>3)*16 + (lane>>4)*8 + (i&7)  -> two 16B loads.
DEV v16bf frag_a(const bf16* base, int ld, int mbase, int k0, int lane) {
  const bf16* p = base + (size_t)(mbase + (lane & 15)) * ld + k0 + ((lane >> 4) << 3);
  return comb(*(const v8bf*)p, *(const v8bf*)(p + 16));
}

// B-matrix fragment (32x16 bf16, KxN) from a row-major (N x K) buffer
// ("column major B" == rows of W^T): lane holds column n = nbase + (lane&15);
// element i maps to k = k0 + (i>>3)*16 + (lane>>4)*8 + (i&7)  -> two 16B loads.
DEV v16bf frag_b(const bf16* base, int ld, int nbase, int k0, int lane) {
  const bf16* p = base + (size_t)(nbase + (lane & 15)) * ld + k0 + ((lane >> 4) << 3);
  return comb(*(const v8bf*)p, *(const v8bf*)(p + 16));
}

DEV v8f wmma_bf16(v16bf a, v16bf b, v8f c) {
  // D(f32 16x16) = A(16x32 bf16) * B(32x16 bf16) + C
  return __builtin_amdgcn_wmma_f32_16x16x32_bf16(false, a, false, b,
                                                 (short)0, c, false, false);
}

// Async global->LDS copy (CDNA5 TDM-adjacent path, tracked by ASYNCcnt).
// Per-lane: LDS[vdst] = MEM[vaddr], 16 bytes. GV mode (SADDR = off/NULL).
DEV void async_copy_b128(const bf16* gsrc, const bf16* lds_dst) {
  unsigned lds = (unsigned)(uintptr_t)lds_dst;          // low 32 bits = LDS offset
  unsigned long long g = (unsigned long long)(uintptr_t)gsrc;
  asm volatile("global_load_async_to_lds_b128 %0, %1, off"
               :: "v"(lds), "v"(g) : "memory");
}
#define WAIT_ASYNC(n) asm volatile("s_wait_asynccnt %0" :: "i"(n) : "memory")

DEV float gelu_exact(float x) {
  return 0.5f * x * (1.0f + erff(x * 0.7071067811865475f));
}

// ---------------------------------------------------------------------------
// Layout transforms
// ---------------------------------------------------------------------------
__global__ void nchw_to_nhwc_k(const float* __restrict__ in, float* __restrict__ outp) {
  size_t i = (size_t)blockIdx.x * 256 + threadIdx.x;
  if (i >= (size_t)16 * 480 * 3136) return;
  int c = (int)(i % 480);
  size_t t = i / 480;
  int hw = (int)(t % 3136);
  int b = (int)(t / 3136);
  outp[i] = in[((size_t)b * 480 + c) * 3136 + hw];
}

__global__ void nhwc_to_nchw_k(const float* __restrict__ X, float* __restrict__ outp) {
  size_t i = (size_t)blockIdx.x * 256 + threadIdx.x;
  if (i >= (size_t)16 * 480 * 3136) return;
  int hw = (int)(i % 3136);
  size_t t = i / 3136;
  int c = (int)(t % 480);
  int b = (int)(t / 480);
  outp[i] = X[((size_t)b * 3136 + hw) * 480 + c];
}

// W (K x N) fp32 row-major -> Wt (N x K) bf16 row-major
__global__ void transpose_w_k(const float* __restrict__ W, bf16* __restrict__ Wt,
                              int K, int N) {
  int idx = blockIdx.x * 256 + threadIdx.x;
  if (idx >= K * N) return;
  int k = idx / N, n = idx - k * N;
  Wt[(size_t)n * K + k] = f2bf(W[idx]);
}

// ---------------------------------------------------------------------------
// LayerNorm1 fused with shift-roll + 7x7 window partition (bf16 out)
// blockIdx.x = windowed row index: win*49 + pos, win = b*64 + wy*8 + wx
// ---------------------------------------------------------------------------
__global__ __launch_bounds__(128) void ln_window_k(const float* __restrict__ X,
                                                   const float* __restrict__ g,
                                                   const float* __restrict__ b,
                                                   bf16* __restrict__ out, int shift) {
  const int tid = threadIdx.x;
  const int row = blockIdx.x;
  const int pos = row % 49, win = row / 49;
  const int batch = win >> 6, wi = win & 63;
  const int wy = wi >> 3, wx = wi & 7;
  const int py = pos / 7, px = pos % 7;
  const int h = (wy * 7 + py + shift) % 56;   // roll(x, -shift)
  const int w = (wx * 7 + px + shift) % 56;
  const float* xr = X + ((size_t)batch * 3136 + h * 56 + w) * 480;

  __shared__ float r1[128], r2[128];
  float s = 0.f, s2 = 0.f;
  for (int c = tid; c < 480; c += 128) { float v = xr[c]; s += v; s2 += v * v; }
  r1[tid] = s; r2[tid] = s2; __syncthreads();
  for (int st = 64; st > 0; st >>= 1) {
    if (tid < st) { r1[tid] += r1[tid + st]; r2[tid] += r2[tid + st]; }
    __syncthreads();
  }
  const float mean = r1[0] * (1.0f / 480.0f);
  const float var = r2[0] * (1.0f / 480.0f) - mean * mean;
  const float inv = rsqrtf(var + 1e-5f);
  bf16* orow = out + (size_t)row * 480;
  for (int c = tid; c < 480; c += 128)
    orow[c] = f2bf((xr[c] - mean) * inv * g[c] + b[c]);
}

// ---------------------------------------------------------------------------
// Residual add (window-reverse + un-roll) + LayerNorm2 (bf16 out, raster order)
// ---------------------------------------------------------------------------
__global__ __launch_bounds__(128) void res_ln2_k(float* X, const float* __restrict__ proj,
                                                 const float* __restrict__ g,
                                                 const float* __restrict__ b,
                                                 bf16* __restrict__ y, int shift) {
  const int tid = threadIdx.x;
  const int row = blockIdx.x;  // b*3136 + h*56 + w
  const int batch = row / 3136, hw = row % 3136;
  const int h = hw / 56, w = hw % 56;
  const int hp = (h - shift + 56) % 56, wp = (w - shift + 56) % 56;  // rolled coords
  const int wrow = (batch * 64 + (hp / 7) * 8 + (wp / 7)) * 49 + (hp % 7) * 7 + (wp % 7);
  float* xr = X + (size_t)row * 480;
  const float* pr = proj + (size_t)wrow * 480;

  float v[4];
  float s = 0.f, s2 = 0.f;
#pragma unroll
  for (int j = 0; j < 4; ++j) {
    int c = tid + j * 128;
    v[j] = 0.f;
    if (c < 480) {
      float t = xr[c] + pr[c];
      v[j] = t; xr[c] = t; s += t; s2 += t * t;
    }
  }
  __shared__ float r1[128], r2[128];
  r1[tid] = s; r2[tid] = s2; __syncthreads();
  for (int st = 64; st > 0; st >>= 1) {
    if (tid < st) { r1[tid] += r1[tid + st]; r2[tid] += r2[tid + st]; }
    __syncthreads();
  }
  const float mean = r1[0] * (1.0f / 480.0f);
  const float var = r2[0] * (1.0f / 480.0f) - mean * mean;
  const float inv = rsqrtf(var + 1e-5f);
  bf16* yr = y + (size_t)row * 480;
#pragma unroll
  for (int j = 0; j < 4; ++j) {
    int c = tid + j * 128;
    if (c < 480) yr[c] = f2bf((v[j] - mean) * inv * g[c] + b[c]);
  }
}

// ---------------------------------------------------------------------------
// Generic WMMA GEMM: out(M x N) = A(M x K, bf16) * Wt(N x K, bf16)^T + bias
// Workgroup: 4 waves / 128 threads, 128(M) x 64(N) tile, K-step 32.
// A tiles double-buffered in LDS via async global->LDS (ASYNCcnt) so the next
// tile streams in while 8 WMMAs/wave run on the current one. B fragments come
// straight from global (weights are tiny and L2-resident on 192MB L2).
// EPI: 0=bias->bf16, 1=bias+GELU->bf16, 2=bias->f32, 3=bias+residual->f32
// M must be a multiple of 128; Wt rows padded to a multiple of 64.
// ---------------------------------------------------------------------------
DEV void issue_tile_async(const bf16* A, int K, int m0, int k0, bf16* buf, int tid) {
#pragma unroll
  for (int j = 0; j < 4; ++j) {
    int ch = tid + j * 128;       // 512 chunks of 16B = 128 rows x 64B
    int r = ch >> 2;
    int off = (ch & 3) << 3;      // bf16 elements (8 per 16B chunk)
    async_copy_b128(A + (size_t)(m0 + r) * K + k0 + off, buf + r * 32 + off);
  }
}

template <int EPI>
__global__ __launch_bounds__(128) void gemm_wmma_k(const bf16* __restrict__ A,
                                                   const bf16* __restrict__ Wt,
                                                   const float* __restrict__ bias,
                                                   const float* res, void* out,
                                                   int M, int N, int K) {
  __shared__ __align__(16) bf16 As[2][128 * 32];
  const int tid = threadIdx.x, lane = tid & 31, wave = tid >> 5;
  const int m0 = blockIdx.y * 128;
  const int nw = blockIdx.x * 64 + wave * 16;   // this wave's 16-wide N strip

  v8f acc[8];
#pragma unroll
  for (int i = 0; i < 8; ++i) acc[i] = vzero8();

  const int nsteps = K >> 5;
  issue_tile_async(A, K, m0, 0, As[0], tid);

  for (int s = 0; s < nsteps; ++s) {
    if (s + 1 < nsteps) {
      issue_tile_async(A, K, m0, (s + 1) << 5, As[(s + 1) & 1], tid);
      WAIT_ASYNC(4);            // tile s complete; tile s+1 (4 issues) in flight
    } else {
      WAIT_ASYNC(0);
    }
    __syncthreads();            // tile s visible to all waves
    const bf16* buf = As[s & 1];
    v16bf bfr = frag_b(Wt, K, nw, s << 5, lane);
#pragma unroll
    for (int mt = 0; mt < 8; ++mt) {
      v16bf af = frag_a(buf, 32, mt * 16, 0, lane);
      acc[mt] = wmma_bf16(af, bfr, acc[mt]);
    }
    __syncthreads();            // done reading buf before it is re-filled
  }

  const int n = nw + (lane & 15);
  if (n >= N) return;
  const float bv = bias[n];
  const int mrow = (lane >> 4) << 3;
#pragma unroll
  for (int mt = 0; mt < 8; ++mt) {
#pragma unroll
    for (int r = 0; r < 8; ++r) {
      const int m = m0 + mt * 16 + mrow + r;
      float v = acc[mt][r] + bv;
      const size_t idx = (size_t)m * N + n;
      if (EPI == 0) ((bf16*)out)[idx] = f2bf(v);
      else if (EPI == 1) ((bf16*)out)[idx] = f2bf(gelu_exact(v));
      else if (EPI == 2) ((float*)out)[idx] = v;
      else ((float*)out)[idx] = v + res[idx];
    }
  }
}

// ---------------------------------------------------------------------------
// Windowed multi-head attention (one workgroup per (window, head)).
// N=49 padded to 64, head_dim padded to 64 (zeros). S = q*k^T via WMMA with
// relative-position bias + shifted-window mask, fp32 softmax, O = P*v via WMMA.
// ---------------------------------------------------------------------------
DEV int mask_region(int c) { return (c < 49) ? 0 : (c < 53 ? 1 : 2); }

__global__ __launch_bounds__(128) void attn_win_k(const bf16* __restrict__ qkv,
                                                  const float* __restrict__ rpb,
                                                  bf16* __restrict__ out,
                                                  int nH, int hd, int shift) {
  __shared__ __align__(16) bf16 qs[64 * 64];
  __shared__ __align__(16) bf16 ks[64 * 64];
  __shared__ __align__(16) bf16 vT[64 * 64];
  __shared__ __align__(16) float Ss[64 * 64];
  const int win = blockIdx.x, head = blockIdx.y;
  const int tid = threadIdx.x, lane = tid & 31, wave = tid >> 5;
  const int wi = win & 63, wy = wi >> 3, wx = wi & 7;
  const float scale = rsqrtf((float)hd);
  const bf16 z = bfzero();

  // Stage q (scaled), k, v^T into LDS, zero-padded to 64x64
  for (int i = tid; i < 64 * 64; i += 128) {
    int t = i >> 6, d = i & 63;
    bf16 qv = z, kv = z, vv = z;
    if (t < 49 && d < hd) {
      const bf16* r = qkv + ((size_t)win * 49 + t) * 1440;
      qv = f2bf(bf2f(r[head * hd + d]) * scale);
      kv = r[480 + head * hd + d];
      vv = r[960 + head * hd + d];
    }
    qs[t * 64 + d] = qv;
    ks[t * 64 + d] = kv;
    vT[d * 64 + t] = vv;
  }
  __syncthreads();

  const int mrow = (lane >> 4) << 3;

  // S = q @ k^T  (+rpb bias, +shift mask), padding -> -1e30
  for (int nt = 0; nt < 4; ++nt) {
    v8f acc = vzero8();
#pragma unroll
    for (int kk = 0; kk < 2; ++kk) {
      v16bf af = frag_a(qs, 64, wave * 16, kk * 32, lane);
      v16bf bfr = frag_b(ks, 64, nt * 16, kk * 32, lane);
      acc = wmma_bf16(af, bfr, acc);
    }
    const int qc = nt * 16 + (lane & 15);
#pragma unroll
    for (int r = 0; r < 8; ++r) {
      const int p = wave * 16 + mrow + r;
      float val = -1.0e30f;
      if (p < 49 && qc < 49) {
        const int py = p / 7, px = p % 7, qy = qc / 7, qx = qc % 7;
        val = acc[r] + rpb[((py - qy + 6) * 13 + (px - qx + 6)) * nH + head];
        if (shift) {
          const int c1 = mask_region(wy * 7 + py) * 3 + mask_region(wx * 7 + px);
          const int c2 = mask_region(wy * 7 + qy) * 3 + mask_region(wx * 7 + qx);
          if (c1 != c2) val -= 100.0f;
        }
      }
      Ss[p * 64 + qc] = val;
    }
  }
  __syncthreads();

  // Row softmax (fp32), write normalized probs as bf16 back into qs
  if (tid < 64) {
    float* row = &Ss[tid * 64];
    float m = -3.0e38f;
    for (int j = 0; j < 64; ++j) m = fmaxf(m, row[j]);
    float sum = 0.f;
    for (int j = 0; j < 64; ++j) { float e = expf(row[j] - m); row[j] = e; sum += e; }
    const float inv = 1.0f / sum;
    for (int j = 0; j < 64; ++j) qs[tid * 64 + j] = f2bf(row[j] * inv);
  }
  __syncthreads();

  // O = P @ v
  const int ntiles = (hd + 15) >> 4;
  for (int nt = 0; nt < ntiles; ++nt) {
    v8f acc = vzero8();
#pragma unroll
    for (int kk = 0; kk < 2; ++kk) {
      v16bf af = frag_a(qs, 64, wave * 16, kk * 32, lane);
      v16bf bfr = frag_b(vT, 64, nt * 16, kk * 32, lane);
      acc = wmma_bf16(af, bfr, acc);
    }
    const int d = nt * 16 + (lane & 15);
    if (d < hd) {
#pragma unroll
      for (int r = 0; r < 8; ++r) {
        const int p = wave * 16 + mrow + r;
        if (p < 49)
          out[((size_t)win * 49 + p) * 480 + head * hd + d] = f2bf(acc[r]);
      }
    }
  }
}

// ---------------------------------------------------------------------------
// Host orchestration
// ---------------------------------------------------------------------------
static constexpr size_t MROWS = 50176;                 // 16*56*56
static constexpr size_t OFF_X = 0;                     // residual stream, fp32
static constexpr size_t SZ_X = MROWS * 480 * 4;
static constexpr size_t OFF_IN = OFF_X + SZ_X;         // ln1-out / ln2-out, bf16
static constexpr size_t SZ_IN = MROWS * 480 * 2;
static constexpr size_t OFF_QKV = OFF_IN + SZ_IN;      // qkv bf16 (also hbuf alias)
static constexpr size_t SZ_QKV = MROWS * 1440 * 2;
static constexpr size_t OFF_ATT = OFF_QKV + SZ_QKV;    // attn out bf16
static constexpr size_t SZ_ATT = MROWS * 480 * 2;
static constexpr size_t OFF_PROJ = OFF_ATT + SZ_ATT;   // proj out fp32
static constexpr size_t SZ_PROJ = MROWS * 480 * 4;
static constexpr size_t OFF_WQKV = OFF_PROJ + SZ_PROJ; // transposed bf16 weights
static constexpr size_t SZ_WQKV = (size_t)1472 * 480 * 2;   // N padded to 64
static constexpr size_t OFF_WPROJ = OFF_WQKV + SZ_WQKV;
static constexpr size_t SZ_WPROJ = (size_t)512 * 480 * 2;
static constexpr size_t OFF_WFC1 = OFF_WPROJ + SZ_WPROJ;
static constexpr size_t SZ_WFC1 = (size_t)1920 * 480 * 2;
static constexpr size_t OFF_WFC2 = OFF_WFC1 + SZ_WFC1;

extern "C" void kernel_launch(void* const* d_in, const int* in_sizes, int n_in,
                              void* d_out, int out_size, void* d_ws, size_t ws_size,
                              hipStream_t stream) {
  (void)in_sizes; (void)n_in; (void)out_size; (void)ws_size;
  const int M = (int)MROWS;
  char* ws = (char*)d_ws;
  float* X = (float*)(ws + OFF_X);
  bf16* XW = (bf16*)(ws + OFF_IN);
  bf16* QKV = (bf16*)(ws + OFF_QKV);
  bf16* ATT = (bf16*)(ws + OFF_ATT);
  float* PROJ = (float*)(ws + OFF_PROJ);
  bf16* HB = (bf16*)(ws + OFF_QKV);   // fc1 hidden aliases dead QKV+ATT regions
  bf16* WQKV = (bf16*)(ws + OFF_WQKV);
  bf16* WPROJ = (bf16*)(ws + OFF_WPROJ);
  bf16* WFC1 = (bf16*)(ws + OFF_WFC1);
  bf16* WFC2 = (bf16*)(ws + OFF_WFC2);

  const size_t total = (size_t)16 * 480 * 3136;
  nchw_to_nhwc_k<<<dim3((unsigned)((total + 255) / 256)), 256, 0, stream>>>(
      (const float*)d_in[0], X);

  const int heads[8] = {8, 8, 16, 16, 32, 32, 32, 32};
  for (int blk = 0; blk < 8; ++blk) {
    const float* P[13];
    for (int j = 0; j < 13; ++j) P[j] = (const float*)d_in[1 + blk * 13 + j];
    // 0:ln1_g 1:ln1_b 2:qkv_w 3:qkv_b 4:rpb 5:proj_w 6:proj_b
    // 7:ln2_g 8:ln2_b 9:fc1_w 10:fc1_b 11:fc2_w 12:fc2_b
    const int nH = heads[blk], hd = 480 / nH;
    const int shift = (blk & 1) ? 3 : 0;

    transpose_w_k<<<(480 * 1440 + 255) / 256, 256, 0, stream>>>(P[2], WQKV, 480, 1440);
    transpose_w_k<<<(480 * 480 + 255) / 256, 256, 0, stream>>>(P[5], WPROJ, 480, 480);
    transpose_w_k<<<(480 * 1920 + 255) / 256, 256, 0, stream>>>(P[9], WFC1, 480, 1920);
    transpose_w_k<<<(1920 * 480 + 255) / 256, 256, 0, stream>>>(P[11], WFC2, 1920, 480);

    ln_window_k<<<M, 128, 0, stream>>>(X, P[0], P[1], XW, shift);
    gemm_wmma_k<0><<<dim3(23, 392), 128, 0, stream>>>(XW, WQKV, P[3], nullptr,
                                                      (void*)QKV, M, 1440, 480);
    attn_win_k<<<dim3(1024, nH), 128, 0, stream>>>(QKV, P[4], ATT, nH, hd, shift);
    gemm_wmma_k<2><<<dim3(8, 392), 128, 0, stream>>>(ATT, WPROJ, P[6], nullptr,
                                                     (void*)PROJ, M, 480, 480);
    res_ln2_k<<<M, 128, 0, stream>>>(X, PROJ, P[7], P[8], XW, shift);
    gemm_wmma_k<1><<<dim3(30, 392), 128, 0, stream>>>(XW, WFC1, P[10], nullptr,
                                                      (void*)HB, M, 1920, 480);
    gemm_wmma_k<3><<<dim3(8, 392), 128, 0, stream>>>(HB, WFC2, P[12], X, (void*)X,
                                                     M, 480, 1920);

    if (blk & 1) {
      float* dst = (float*)d_out + (size_t)(blk >> 1) * total;
      nhwc_to_nchw_k<<<dim3((unsigned)((total + 255) / 256)), 256, 0, stream>>>(X, dst);
    }
  }
}
